// WindowAttention_48352741818780
// MI455X (gfx1250) — matrix operand order repaired and checked by reading, not compiled, not run on hardware
//
#include <hip/hip_runtime.h>
#include <hip/hip_bf16.h>

typedef _Float16 half_t;
typedef __attribute__((ext_vector_type(16))) _Float16 v16h;
typedef __attribute__((ext_vector_type(8)))  _Float16 v8h;
typedef __attribute__((ext_vector_type(8)))  float    v8f;

#define N_TOK   49
#define DIM_    384
#define HEADS_  12
#define HD_     32
#define BATCH_  2048
#define SCALE_Q 0.17677669529663687f

// ---- LDS layout (units: half elements), all regions 16B aligned ----
#define XH_LD    392                      // x / attn-out tile, 64 rows
#define XH_HALFS (64 * XH_LD)             // 25088
#define QK_OFF   XH_HALFS
#define QK_LD    776                      // q|k (cols 0..767), 64 rows
#define QK_HALFS (64 * QK_LD)             // 49664
#define VT_OFF   (QK_OFF + QK_HALFS)      // 74752
#define VT_LD    72                       // v transposed: [384 d][64 tok]
#define VT_HALFS (384 * VT_LD)            // 27648
#define PR_OFF   (VT_OFF + VT_HALFS)      // 102400
#define PR_LD    72                       // per-wave prob tile 16x64 (pad)
#define PR_WAVE  (16 * PR_LD)             // 1152
#define SMEM_HALFS (PR_OFF + 8 * PR_WAVE) // 111616
#define SMEM_BYTES (SMEM_HALFS * 2)       // 223232 < 320KB

// ---- workspace layout (bytes) ----
#define WS_RPB_OFF 0                       // 12*49*49 f32 = 115248 B
#define WS_QWH_OFF (128 * 1024)            // 1152*384 f16 = 884736 B
#define WS_PWH_OFF (1024 * 1024)           // 384*384  f16 = 294912 B

#define QKV_W_ELEMS (1152 * 384)
#define PROJ_W_ELEMS (384 * 384)
#define RPB_ELEMS (HEADS_ * N_TOK * N_TOK)

// A fragment (f16 16x32): lane&15 -> row, lanes<16 hold K {k0..k0+7, k0+16..k0+23},
// lanes>=16 hold K {k0+8..k0+15, k0+24..k0+31}  (ISA 7.12.2 16-bit A table)
__device__ __forceinline__ v16h load_a16(const half_t* base, int ld, int row0,
                                         int k0, int lane) {
  int r  = row0 + (lane & 15);
  int kb = k0 + ((lane & 16) ? 8 : 0);
  const half_t* p = base + r * ld + kb;
  union { v16h v; v8h h[2]; } u;
  u.h[0] = *(const v8h*)p;          // K kb .. kb+7
  u.h[1] = *(const v8h*)(p + 16);   // K kb+16 .. kb+23
  return u.v;
}

// B fragment (f16 32x16) where B[k][n] = src[n*ld + k] (src holds columns as rows):
// lane&15 -> n, lanes<16 hold K {k0..k0+7, k0+16..k0+23}, lanes>=16 hold K {k0+8..k0+15, k0+24..k0+31}
__device__ __forceinline__ v16h load_bt(const half_t* base, int ld, int n0,
                                        int k0, int lane) {
  int n  = n0 + (lane & 15);
  int kb = k0 + ((lane & 16) ? 8 : 0);
  const half_t* p = base + n * ld + kb;
  union { v16h v; v8h h[2]; } u;
  u.h[0] = *(const v8h*)p;          // K kb .. kb+7
  u.h[1] = *(const v8h*)(p + 16);   // K kb+16 .. kb+23
  return u.v;
}

__device__ __forceinline__ v8f wmma_f16(v16h a, v16h b, v8f c) {
  return __builtin_amdgcn_wmma_f32_16x16x32_f16(false, a, false, b, (short)0, c,
                                                false, false);
}

// ---------- prep: f32->f16 weights + rpb gather materialization ----------
__global__ void swin_prep_kernel(const float* __restrict__ qkv_w,
                                 const float* __restrict__ proj_w,
                                 const float* __restrict__ bias_table,
                                 const int*   __restrict__ rel_idx,
                                 half_t* __restrict__ qwh,
                                 half_t* __restrict__ pwh,
                                 float*  __restrict__ rpb) {
  int idx = blockIdx.x * blockDim.x + threadIdx.x;
  int stride = gridDim.x * blockDim.x;
  for (int i = idx; i < QKV_W_ELEMS; i += stride) qwh[i] = (half_t)qkv_w[i];
  for (int i = idx; i < PROJ_W_ELEMS; i += stride) pwh[i] = (half_t)proj_w[i];
  for (int i = idx; i < RPB_ELEMS; i += stride) {
    int h = i / (N_TOK * N_TOK);
    int qk = i - h * (N_TOK * N_TOK);
    rpb[i] = bias_table[rel_idx[qk] * HEADS_ + h];
  }
}

// ---------- fused window attention: one block per window ----------
__global__ __launch_bounds__(256, 1)
void swin_fused_kernel(const float* __restrict__ x,
                       const float* __restrict__ qkv_b,
                       const float* __restrict__ proj_b,
                       const float* __restrict__ mask,
                       const float* __restrict__ rpb,
                       const half_t* __restrict__ qwh,
                       const half_t* __restrict__ pwh,
                       float* __restrict__ out) {
  extern __shared__ __align__(16) half_t smem[];
  half_t* xh = smem;                 // x tile -> later attn-out tile (f16)
  half_t* qk = smem + QK_OFF;        // q (cols 0..383) | k (cols 384..767)
  half_t* vt = smem + VT_OFF;        // v transposed [d][token]
  const int tid  = threadIdx.x;
  const int wave = tid >> 5;
  const int lane = tid & 31;
  half_t* pr = smem + PR_OFF + wave * PR_WAVE;
  const int b = blockIdx.x;
  const int kc = lane & 15;
  const int rhalf = (lane & 16) ? 8 : 0;

  // ---- phase 0: stage x (f32 -> f16), zero the 15 pad rows ----
  {
    const float* xb = x + (size_t)b * (N_TOK * DIM_);
    for (int i = tid; i < N_TOK * DIM_; i += 256) {
      int r = i / DIM_, c = i - r * DIM_;
      xh[r * XH_LD + c] = (half_t)xb[i];
    }
    for (int i = tid; i < 15 * XH_LD; i += 256)
      xh[N_TOK * XH_LD + i] = (half_t)0.f;
  }
  __syncthreads();

  // ---- phase 1: qkv = x @ qkv_w^T + qkv_b  (M=64, N=1152, K=384) ----
  for (int j = wave; j < 288; j += 8) {
    int mt = j & 3, nt = j >> 2;
    int m0 = mt * 16, n0 = nt * 16;
    v8f acc = {};
#pragma unroll
    for (int kkk = 0; kkk < 12; ++kkk) {
      v16h a  = load_a16(xh, XH_LD, m0, kkk * 32, lane);
      v16h bm = load_bt(qwh, DIM_, n0, kkk * 32, lane);
      acc = wmma_f16(a, bm, acc);
    }
    int n = n0 + kc;
    float bias = qkv_b[n];
#pragma unroll
    for (int r = 0; r < 8; ++r) {
      int m = m0 + r + rhalf;
      float v = acc[r] + bias;
      if (n < 2 * DIM_) qk[m * QK_LD + n] = (half_t)v;          // q | k
      else              vt[(n - 2 * DIM_) * VT_LD + m] = (half_t)v; // v^T
    }
  }
  __syncthreads();

  // ---- phase 2: attention, one (head, q-row-tile) job per wave-iter ----
  {
    const float* mk = mask + (size_t)(b & 63) * (N_TOK * N_TOK);
    for (int j = wave; j < 48; j += 8) {
      int h = j >> 2, mt = j & 3;
      int m0 = mt * 16;
      const float* rp = rpb + (size_t)h * (N_TOK * N_TOK);

      // scores = q @ k^T  (K = 32, one WMMA per 16x16 tile)
      v16h aq = load_a16(qk, QK_LD, m0, h * HD_, lane);
      v8f c[4];
#pragma unroll
      for (int t = 0; t < 4; ++t) {
        v8f z = {};
        v16h bk = load_bt(qk + DIM_ + h * HD_, QK_LD, t * 16, 0, lane);
        c[t] = wmma_f16(aq, bk, z);
      }

      // scale + rpb + mask, register-resident softmax over k (49 valid cols)
#pragma unroll
      for (int r = 0; r < 8; ++r) {
        int q = m0 + r + rhalf;
        float mx = -1e30f;
#pragma unroll
        for (int t = 0; t < 4; ++t) {
          int k = t * 16 + kc;
          float v = -1e30f;
          if (q < N_TOK && k < N_TOK)
            v = c[t][r] * SCALE_Q + rp[q * N_TOK + k] + mk[q * N_TOK + k];
          c[t][r] = v;
          mx = fmaxf(mx, v);
        }
#pragma unroll
        for (int s = 1; s < 16; s <<= 1) mx = fmaxf(mx, __shfl_xor(mx, s, 32));
        float sum = 0.f;
#pragma unroll
        for (int t = 0; t < 4; ++t) {
          float ev = __expf(c[t][r] - mx);   // pad cols underflow to 0
          c[t][r] = ev;
          sum += ev;
        }
#pragma unroll
        for (int s = 1; s < 16; s <<= 1) sum += __shfl_xor(sum, s, 32);
        float inv = 1.f / sum;
#pragma unroll
        for (int t = 0; t < 4; ++t)
          pr[(r + rhalf) * PR_LD + t * 16 + kc] = (half_t)(c[t][r] * inv);
      }

      // out = probs @ v   (K = 64 zero-padded, N = 32)
      v16h ap0 = load_a16(pr, PR_LD, 0, 0, lane);
      v16h ap1 = load_a16(pr, PR_LD, 0, 32, lane);
#pragma unroll
      for (int t2 = 0; t2 < 2; ++t2) {
        v8f acc = {};
        v16h bv0 = load_bt(vt, VT_LD, h * HD_ + t2 * 16, 0, lane);
        acc = wmma_f16(ap0, bv0, acc);
        v16h bv1 = load_bt(vt, VT_LD, h * HD_ + t2 * 16, 32, lane);
        acc = wmma_f16(ap1, bv1, acc);
#pragma unroll
        for (int r = 0; r < 8; ++r)
          xh[(m0 + r + rhalf) * XH_LD + h * HD_ + t2 * 16 + kc] =
              (half_t)acc[r];
      }
    }
  }
  __syncthreads();

  // ---- phase 3: out = attn_out @ proj_w^T + proj_b  (M=64, N=384, K=384) ----
  for (int j = wave; j < 96; j += 8) {
    int mt = j & 3, nt = j >> 2;
    int m0 = mt * 16, n0 = nt * 16;
    v8f acc = {};
#pragma unroll
    for (int kkk = 0; kkk < 12; ++kkk) {
      v16h a  = load_a16(xh, XH_LD, m0, kkk * 32, lane);
      v16h bm = load_bt(pwh, DIM_, n0, kkk * 32, lane);
      acc = wmma_f16(a, bm, acc);
    }
    int n = n0 + kc;
    float bias = proj_b[n];
#pragma unroll
    for (int r = 0; r < 8; ++r) {
      int m = m0 + r + rhalf;
      if (m < N_TOK)
        out[((size_t)b * N_TOK + m) * DIM_ + n] = acc[r] + bias;
    }
  }
}

extern "C" void kernel_launch(void* const* d_in, const int* in_sizes, int n_in,
                              void* d_out, int out_size, void* d_ws,
                              size_t ws_size, hipStream_t stream) {
  (void)in_sizes; (void)n_in; (void)out_size; (void)ws_size;
  const float* x    = (const float*)d_in[0];
  const float* qkvw = (const float*)d_in[1];
  const float* qkvb = (const float*)d_in[2];
  const float* pw   = (const float*)d_in[3];
  const float* pb   = (const float*)d_in[4];
  const float* bt   = (const float*)d_in[5];
  const float* mask = (const float*)d_in[6];
  const int*   ridx = (const int*)d_in[7];

  char* ws = (char*)d_ws;
  float*  rpb = (float*)(ws + WS_RPB_OFF);
  half_t* qwh = (half_t*)(ws + WS_QWH_OFF);
  half_t* pwh = (half_t*)(ws + WS_PWH_OFF);

  swin_prep_kernel<<<432, 256, 0, stream>>>(qkvw, pw, bt, ridx, qwh, pwh, rpb);

  static_assert(SMEM_BYTES <= 320 * 1024, "LDS budget");
  hipFuncSetAttribute((const void*)swin_fused_kernel,
                      hipFuncAttributeMaxDynamicSharedMemorySize, SMEM_BYTES);
  swin_fused_kernel<<<BATCH_, 256, SMEM_BYTES, stream>>>(
      x, qkvb, pb, mask, rpb, qwh, pwh, (float*)d_out);
}
